// GraphLearningLayer_29343216566667
// MI455X (gfx1250) — matrix operand order repaired and checked by reading, not compiled, not run on hardware
//
#include <hip/hip_runtime.h>
#include <hip/hip_bf16.h>

typedef __attribute__((ext_vector_type(16))) _Float16 v16h;
typedef __attribute__((ext_vector_type(8)))  float    v8f;

#define NN    8192
#define DIMD  64
#define KSEL  32

// ---------------------------------------------------------------------------
// fast tanh: prefer gfx1250 native v_tanh_f32 (TRANS op) if clang exposes it.
// ---------------------------------------------------------------------------
__device__ __forceinline__ float fast_tanh(float z) {
#if __has_builtin(__builtin_amdgcn_tanhf)
  return __builtin_amdgcn_tanhf(z);
#elif __has_builtin(__builtin_amdgcn_tanh_f32)
  return __builtin_amdgcn_tanh_f32(z);
#else
  z = fminf(15.0f, fmaxf(-15.0f, z));
  const float e = __expf(-2.0f * z);
  return __fdividef(1.0f - e, 1.0f + e);   // v_rcp_f32-based fast divide
#endif
}

// ---------------------------------------------------------------------------
// Fragment loaders for V_WMMA_F32_16X16X32_F16 (wave32).
// A (16x32, MxK): lane holds row M=lane&15; halves 0..15 = K=kb+hi*16+0..15
// where hi = lane>>4.
// ---------------------------------------------------------------------------
__device__ __forceinline__ v16h load_a_frag(const _Float16* __restrict__ M,
                                            int row0, int kb, int lane) {
  const int hi  = lane >> 4;
  const int row = row0 + (lane & 15);
  const _Float16* p = M + (size_t)row * DIMD + kb + hi * 16;
  union { v16h v; uint4 u[2]; } t;
  t.u[0] = *(const uint4*)(p);
  t.u[1] = *(const uint4*)(p + 8);
  return t.v;
}

// B (32x16, KxN) with B[k][n] = Msrc[j0+n][k]. Lane L holds column N=L&15,
// k = kb + (L>>4)*16 + 0..15 -> contiguous 32B in row-major Msrc.
__device__ __forceinline__ v16h load_b_frag(const _Float16* __restrict__ M,
                                            int row0, int kb, int lane) {
  const int row = row0 + (lane & 15);
  const _Float16* p = M + (size_t)row * DIMD + kb + (lane >> 4) * 16;
  union { v16h v; uint4 u[2]; } t;
  t.u[0] = *(const uint4*)(p);
  t.u[1] = *(const uint4*)(p + 8);
  return t.v;
}

__device__ __forceinline__ v8f wmma_f16(v16h a, v16h b, v8f c) {
  return __builtin_amdgcn_wmma_f32_16x16x32_f16(false, a, false, b,
                                                (short)0, c, false, false);
}

// ---------------------------------------------------------------------------
// Stage 1: M1 = tanh(3*(emb1[x] @ W1^T + b1)) -> f16
//          M2 = tanh(...)                     -> f16, plus M2n = -M2 (f16).
// M2n lets stage 2 fuse M1@M2^T - M2@M1^T into ONE accumulator chain:
//          A = M1@M2^T + (-M2)@M1^T    (f16 negation is exact)
// ---------------------------------------------------------------------------
__global__ __launch_bounds__(256) void prep_kernel(
    const int* __restrict__ x,
    const float* __restrict__ emb1, const float* __restrict__ emb2,
    const float* __restrict__ W1, const float* __restrict__ b1,
    const float* __restrict__ W2, const float* __restrict__ b2,
    _Float16* __restrict__ M1h, _Float16* __restrict__ M2h,
    _Float16* __restrict__ M2n) {
  __shared__ float sW1[DIMD * DIMD];
  __shared__ float sW2[DIMD * DIMD];
  __shared__ float se1[4 * DIMD];
  __shared__ float se2[4 * DIMD];
  const int tid = threadIdx.x;
  for (int i = tid; i < DIMD * DIMD; i += 256) { sW1[i] = W1[i]; sW2[i] = W2[i]; }
  const int local = tid >> 6;
  const int d     = tid & 63;
  const int node  = blockIdx.x * 4 + local;
  const int g     = x[node];
  se1[local * DIMD + d] = emb1[(size_t)g * DIMD + d];
  se2[local * DIMD + d] = emb2[(size_t)g * DIMD + d];
  __syncthreads();
  float acc1 = b1[d], acc2 = b2[d];
#pragma unroll 8
  for (int k = 0; k < DIMD; ++k) {
    acc1 += se1[local * DIMD + k] * sW1[d * DIMD + k];
    acc2 += se2[local * DIMD + k] * sW2[d * DIMD + k];
  }
  const float t1 = fast_tanh(3.0f * acc1);
  const float t2 = fast_tanh(3.0f * acc2);
  M1h[(size_t)node * DIMD + d] = (_Float16)t1;
  M2h[(size_t)node * DIMD + d] = (_Float16)t2;
  M2n[(size_t)node * DIMD + d] = (_Float16)(-t2);
}

// ---------------------------------------------------------------------------
// Stage 2: A = relu(tanh(3*(M1@M2^T + (-M2)@M1^T)))
// grid (256, 8), 8 waves/block. Each wave owns TWO i-tiles (32 rows),
// iterates 8 j-tiles with software-pipelined B-fragment prefetch.
// Per j step: 8 back-to-back v_wmma_f32_16x16x32_f16 into just 2 accumulators,
// no subtraction pass, next step's loads in flight during the WMMAs.
// ---------------------------------------------------------------------------
__global__ __launch_bounds__(256) void score_kernel(
    const _Float16* __restrict__ M1h, const _Float16* __restrict__ M2h,
    const _Float16* __restrict__ M2n, float* __restrict__ out) {
  const int lane = threadIdx.x & 31;
  const int wave = threadIdx.x >> 5;
  const int i0   = blockIdx.x * 32;                 // i-tiles: i0, i0+16

  // persistent A-operands for both i-tiles (live in VGPRs across the loop)
  const v16h a1k0_0 = load_a_frag(M1h, i0,      0,  lane);
  const v16h a1k1_0 = load_a_frag(M1h, i0,      32, lane);
  const v16h a2k0_0 = load_a_frag(M2n, i0,      0,  lane);   // negated M2
  const v16h a2k1_0 = load_a_frag(M2n, i0,      32, lane);
  const v16h a1k0_1 = load_a_frag(M1h, i0 + 16, 0,  lane);
  const v16h a1k1_1 = load_a_frag(M1h, i0 + 16, 32, lane);
  const v16h a2k0_1 = load_a_frag(M2n, i0 + 16, 0,  lane);
  const v16h a2k1_1 = load_a_frag(M2n, i0 + 16, 32, lane);

  const int hi = lane >> 4;

  int jt = blockIdx.y * 8 + wave;                   // stride-64 over 512 tiles
  v16h b2k0 = load_b_frag(M2h, jt * 16, 0,  lane);
  v16h b2k1 = load_b_frag(M2h, jt * 16, 32, lane);
  v16h b1k0 = load_b_frag(M1h, jt * 16, 0,  lane);
  v16h b1k1 = load_b_frag(M1h, jt * 16, 32, lane);

#pragma unroll 1
  for (int s = 0; s < 8; ++s) {
    const int j0 = jt * 16;
    jt = (jt + 64) & (NN / 16 - 1);                 // wraps on last iter (harmless)

    // prefetch next j-tile B fragments (only need to be in flight below)
    const v16h nb2k0 = load_b_frag(M2h, jt * 16, 0,  lane);
    const v16h nb2k1 = load_b_frag(M2h, jt * 16, 32, lane);
    const v16h nb1k0 = load_b_frag(M1h, jt * 16, 0,  lane);
    const v16h nb1k1 = load_b_frag(M1h, jt * 16, 32, lane);

    v8f ca = {}, cb = {};
    ca = wmma_f16(a1k0_0, b2k0, ca);  ca = wmma_f16(a1k1_0, b2k1, ca);
    ca = wmma_f16(a2k0_0, b1k0, ca);  ca = wmma_f16(a2k1_0, b1k1, ca);
    cb = wmma_f16(a1k0_1, b2k0, cb);  cb = wmma_f16(a1k1_1, b2k1, cb);
    cb = wmma_f16(a2k0_1, b1k0, cb);  cb = wmma_f16(a2k1_1, b1k1, cb);

    const int col = j0 + (lane & 15);
#pragma unroll
    for (int v = 0; v < 8; ++v) {
      const float ta = fmaxf(fast_tanh(3.0f * ca[v]), 0.0f);   // relu∘tanh
      const float tb = fmaxf(fast_tanh(3.0f * cb[v]), 0.0f);
      out[(size_t)(i0 + hi * 8 + v) * NN + col]      = ta;
      out[(size_t)(i0 + 16 + hi * 8 + v) * NN + col] = tb;
    }

    b2k0 = nb2k0; b2k1 = nb2k1; b1k0 = nb1k0; b1k1 = nb1k1;
  }
}

// ---------------------------------------------------------------------------
// Stage 3: per-row top-32 masking, in place on d_out. One block per row.
// Wave32 shuffle argmax + one 8-entry cross-wave LDS step per round.
// ---------------------------------------------------------------------------
__global__ __launch_bounds__(256) void topk_kernel(float* __restrict__ out) {
  __shared__ float    vals[NN];        // 32 KB row cache
  __shared__ unsigned selm[NN / 32];   // selection bitmask
  __shared__ float    rv[8];
  __shared__ int      ri[8];

  const int    tid = threadIdx.x;
  const int    wid = tid >> 5;
  const int    lid = tid & 31;
  const size_t row = blockIdx.x;
  float* __restrict__ rowp = out + row * NN;

  float orig[32];
#pragma unroll
  for (int i = 0; i < 32; ++i) {
    const int idx = tid + i * 256;
    const float v = rowp[idx];
    orig[i]   = v;
    vals[idx] = v;
  }
  selm[tid] = 0u;
  __syncthreads();

  for (int k = 0; k < KSEL; ++k) {
    float bv = -1.0f;
    int   bi = tid;
#pragma unroll
    for (int i = 0; i < 32; ++i) {
      const int idx = tid + i * 256;
      const float v = vals[idx];
      if (v > bv) { bv = v; bi = idx; }
    }
#pragma unroll
    for (int off = 16; off > 0; off >>= 1) {        // wave32 argmax
      const float ov = __shfl_xor(bv, off, 32);
      const int   oi = __shfl_xor(bi, off, 32);
      if (ov > bv) { bv = ov; bi = oi; }
    }
    if (lid == 0) { rv[wid] = bv; ri[wid] = bi; }
    __syncthreads();
    if (tid == 0) {
      float gv = rv[0]; int gi = ri[0];
#pragma unroll
      for (int w = 1; w < 8; ++w)
        if (rv[w] > gv) { gv = rv[w]; gi = ri[w]; }
      vals[gi] = -2.0f;                      // remove from later rounds
      selm[gi >> 5] |= (1u << (gi & 31));    // mark selected
    }
    __syncthreads();
  }

#pragma unroll
  for (int i = 0; i < 32; ++i) {
    const int idx = tid + i * 256;
    const bool sel = (selm[idx >> 5] >> (idx & 31)) & 1u;
    rowp[idx] = sel ? orig[i] : 0.0f;
  }
}

// ---------------------------------------------------------------------------
extern "C" void kernel_launch(void* const* d_in, const int* in_sizes, int n_in,
                              void* d_out, int out_size, void* d_ws, size_t ws_size,
                              hipStream_t stream) {
  (void)in_sizes; (void)n_in; (void)out_size; (void)ws_size;
  const int*   x    = (const int*)d_in[0];
  const float* emb1 = (const float*)d_in[1];
  const float* emb2 = (const float*)d_in[2];
  const float* W1   = (const float*)d_in[3];
  const float* b1   = (const float*)d_in[4];
  const float* W2   = (const float*)d_in[5];
  const float* b2   = (const float*)d_in[6];
  float*       out  = (float*)d_out;

  const size_t mbytes = (size_t)NN * DIMD * sizeof(_Float16);   // 1 MB each
  _Float16* M1h = (_Float16*)d_ws;
  _Float16* M2h = (_Float16*)((char*)d_ws + mbytes);
  _Float16* M2n = (_Float16*)((char*)d_ws + 2 * mbytes);

  prep_kernel<<<NN / 4, 256, 0, stream>>>(x, emb1, emb2, W1, b1, W2, b2,
                                          M1h, M2h, M2n);
  score_kernel<<<dim3(NN / 32, 8), 256, 0, stream>>>(M1h, M2h, M2n, out);
  topk_kernel<<<NN, 256, 0, stream>>>(out);
}
